// SiameseGNN_GIN_46325517254968
// MI455X (gfx1250) — compile-verified, matrix operand order checked by reading, not run on hardware
//
#include <hip/hip_runtime.h>
#include <hip/hip_bf16.h>
#include <math.h>

#define NNODES 20000
#define NEDGES 640000
#define FINDIM 256
#define HDIM   512
#define TOPK   1000
#define SORTN  32768

#define USE_ASYNC_LDS 1

typedef __attribute__((ext_vector_type(16))) __bf16 v16bf;
typedef __attribute__((ext_vector_type(8)))  float  v8f;

// ---------- bf16 helpers ----------
__device__ __forceinline__ unsigned short f2bf(float f) {
  unsigned int u = __float_as_uint(f);
  return (unsigned short)((u + 0x7fffu + ((u >> 16) & 1u)) >> 16);
}
__device__ __forceinline__ unsigned int pack_bf2(float a, float b) {
  return ((unsigned int)f2bf(b) << 16) | (unsigned int)f2bf(a);
}
__device__ __forceinline__ float bf_lo(unsigned int u) { return __uint_as_float(u << 16); }
__device__ __forceinline__ float bf_hi(unsigned int u) { return __uint_as_float(u & 0xffff0000u); }

// 16-byte global -> LDS copy (async on CDNA5)
__device__ __forceinline__ void copy16_g2l(const unsigned short* gsrc, unsigned short* ldst) {
#if USE_ASYNC_LDS
  unsigned int loff = (unsigned int)(unsigned long long)ldst;   // LDS byte offset = addr[31:0]
  unsigned long long ga = (unsigned long long)gsrc;
  asm volatile("global_load_async_to_lds_b128 %0, %1, off" :: "v"(loff), "v"(ga) : "memory");
#else
  *(uint4*)ldst = *(const uint4*)gsrc;
#endif
}
__device__ __forceinline__ void staging_fence() {
#if USE_ASYNC_LDS
  asm volatile("s_wait_asynccnt 0x0" ::: "memory");
#endif
  __syncthreads();
}

// ---------- weight convert + transpose: W[K][Nout] f32 -> WT[Nout][K] bf16 ----------
__global__ void wconv_kernel(const float* __restrict__ w, unsigned short* __restrict__ wt,
                             int K, int Nout) {
  int tid = blockIdx.x * blockDim.x + threadIdx.x;
  int total = K * Nout;
  if (tid >= total) return;
  int n = tid / K;
  int k = tid - n * K;
  wt[(size_t)n * K + k] = f2bf(w[(size_t)k * Nout + n]);
}

// ---------- elementwise f32 -> bf16 ----------
__global__ void xconv_kernel(const float* __restrict__ in, unsigned short* __restrict__ outp, int n) {
  int i = blockIdx.x * blockDim.x + threadIdx.x;
  if (i < n) outp[i] = f2bf(in[i]);
}

// ---------- zero fill ----------
__global__ void zero_kernel(float* __restrict__ p, int n) {
  int i = blockIdx.x * blockDim.x + threadIdx.x;
  if (i < n) p[i] = 0.0f;
}

// ---------- edge scatter-add (bf16 source, f32 atomic accumulate) ----------
__global__ void scatter_bf16_kernel(const unsigned short* __restrict__ X,
                                    const int* __restrict__ src, const int* __restrict__ dst,
                                    float* __restrict__ AGG, int nedge, int F) {
  int tid = blockIdx.x * blockDim.x + threadIdx.x;
  int chunks = F >> 3;                 // 8 bf16 per thread
  int total = nedge * chunks;
  if (tid >= total) return;
  int e = tid / chunks;
  int c = (tid - e * chunks) << 3;
  int s = src[e];
  int d = dst[e];
  uint4 r = *(const uint4*)(X + (size_t)s * F + c);
  float* o = AGG + (size_t)d * F + c;
  atomicAdd(o + 0, bf_lo(r.x)); atomicAdd(o + 1, bf_hi(r.x));
  atomicAdd(o + 2, bf_lo(r.y)); atomicAdd(o + 3, bf_hi(r.y));
  atomicAdd(o + 4, bf_lo(r.z)); atomicAdd(o + 5, bf_hi(r.z));
  atomicAdd(o + 6, bf_lo(r.w)); atomicAdd(o + 7, bf_hi(r.w));
}

// ---------- WMMA GEMM: Cbf = relu((A [+A2]) @ WT^T + bias), bf16 in / bf16 out ----------
// Abf: [M][K] bf16, A2: optional [M][K] f32 residual, WT: [Nout][K] bf16.
// Block tile 128x128, 8 waves (4M x 2N), wave tile 32x64 => 2x4 WMMA, BLK_K=32.
__global__ __launch_bounds__(256)
void gemm_bf16_kernel(const unsigned short* __restrict__ Abf, const float* __restrict__ A2,
                      const unsigned short* __restrict__ WT, const float* __restrict__ bias,
                      unsigned short* __restrict__ Cbf, int M, int K, int Nout, int do_relu) {
  __shared__ unsigned short lA[128 * 32];
  __shared__ unsigned short lB[128 * 32];

  const int t    = threadIdx.x;
  const int lane = t & 31;
  const int wid  = t >> 5;
  const int wm   = wid & 3;     // 0..3 -> 32 rows each
  const int wn   = wid >> 2;    // 0..1 -> 64 cols each
  const int lr   = lane & 15;
  const int kh   = lane >> 4;

  const int blockM = blockIdx.x * 128;
  const int blockN = blockIdx.y * 128;

  v8f acc[2][4] = {};

  const int seg  = t >> 1;      // 0..127 : A row / B col
  const int half = t & 1;       // 16 K-elements each
  const int grow = blockM + seg;

  for (int k0 = 0; k0 < K; k0 += 32) {
    // ---- stage B tile: WT[col][k] contiguous in K -> direct (async) copy ----
    {
      const unsigned short* wp = WT + (size_t)(blockN + seg) * K + k0 + half * 16;
      unsigned short* lp = &lB[seg * 32 + half * 16];
      copy16_g2l(wp + 0, lp + 0);
      copy16_g2l(wp + 8, lp + 8);
    }
    // ---- stage A tile ----
    if (grow < M) {
      const unsigned short* ap = Abf + (size_t)grow * K + k0 + half * 16;
      unsigned short* lp = &lA[seg * 32 + half * 16];
      if (A2 == nullptr) {
        copy16_g2l(ap + 0, lp + 0);
        copy16_g2l(ap + 8, lp + 8);
      } else {
        uint4 r0 = *(const uint4*)(ap + 0);
        uint4 r1 = *(const uint4*)(ap + 8);
        const float4* cp = (const float4*)(A2 + (size_t)grow * K + k0 + half * 16);
        float4 g0 = cp[0], g1 = cp[1], g2 = cp[2], g3 = cp[3];
        __builtin_prefetch(ap + 32, 0, 1);  // next K-tile
        uint4 p0, p1;
        p0.x = pack_bf2(bf_lo(r0.x) + g0.x, bf_hi(r0.x) + g0.y);
        p0.y = pack_bf2(bf_lo(r0.y) + g0.z, bf_hi(r0.y) + g0.w);
        p0.z = pack_bf2(bf_lo(r0.z) + g1.x, bf_hi(r0.z) + g1.y);
        p0.w = pack_bf2(bf_lo(r0.w) + g1.z, bf_hi(r0.w) + g1.w);
        p1.x = pack_bf2(bf_lo(r1.x) + g2.x, bf_hi(r1.x) + g2.y);
        p1.y = pack_bf2(bf_lo(r1.y) + g2.z, bf_hi(r1.y) + g2.w);
        p1.z = pack_bf2(bf_lo(r1.z) + g3.x, bf_hi(r1.z) + g3.y);
        p1.w = pack_bf2(bf_lo(r1.w) + g3.z, bf_hi(r1.w) + g3.w);
        uint4* dp = (uint4*)lp;
        dp[0] = p0;
        dp[1] = p1;
      }
    }
    staging_fence();

    // ---- WMMA compute ----
    union Frag { uint4 u[2]; v16bf v; };
    Frag fa[2], fb[4];
#pragma unroll
    for (int i = 0; i < 2; ++i) {
      int r = wm * 32 + i * 16 + lr;
      fa[i].u[0] = *(const uint4*)&lA[r * 32 + kh * 8];       // K = 8*kh..+7
      fa[i].u[1] = *(const uint4*)&lA[r * 32 + 16 + kh * 8];  // K = 16+8*kh..+7
    }
#pragma unroll
    for (int j = 0; j < 4; ++j) {
      int c = wn * 64 + j * 16 + lr;
      fb[j].u[0] = *(const uint4*)&lB[c * 32 + kh * 8];
      fb[j].u[1] = *(const uint4*)&lB[c * 32 + 16 + kh * 8];
    }
#pragma unroll
    for (int i = 0; i < 2; ++i)
#pragma unroll
      for (int j = 0; j < 4; ++j)
        acc[i][j] = __builtin_amdgcn_wmma_f32_16x16x32_bf16(
            false, fa[i].v, false, fb[j].v, (short)0, acc[i][j], false, false);
    __syncthreads();
  }

  // ---- epilogue: bias + relu, bf16 store ----
#pragma unroll
  for (int i = 0; i < 2; ++i) {
#pragma unroll
    for (int j = 0; j < 4; ++j) {
      int col = blockN + wn * 64 + j * 16 + lr;
      float bv = bias[col];
#pragma unroll
      for (int r = 0; r < 8; ++r) {
        int m = blockM + wm * 32 + i * 16 + kh * 8 + r;
        if (m < M) {
          float v = acc[i][j][r] + bv;
          if (do_relu) v = fmaxf(v, 0.0f);
          Cbf[(size_t)m * Nout + col] = f2bf(v);
        }
      }
    }
  }
}

// ---------- final linear projection (bf16 activations) to 2 dims ----------
__global__ void lin_kernel(const unsigned short* __restrict__ Y, const float* __restrict__ lw,
                           const float* __restrict__ lb, float* __restrict__ O, int n) {
  int i = blockIdx.x * blockDim.x + threadIdx.x;
  if (i >= n) return;
  const unsigned short* y = Y + (size_t)i * HDIM;
  float a0 = lb[0], a1 = lb[1];
  for (int k = 0; k < HDIM; ++k) {
    float v = __uint_as_float(((unsigned int)y[k]) << 16);
    a0 += v * lw[2 * k + 0];
    a1 += v * lw[2 * k + 1];
  }
  O[2 * i + 0] = a0;
  O[2 * i + 1] = a1;
}

// ---------- pairwise distance into padded sort buffer ----------
__global__ void dist_kernel(const float* __restrict__ o1, const float* __restrict__ o2,
                            float* __restrict__ sb, int n) {
  int i = blockIdx.x * blockDim.x + threadIdx.x;
  if (i >= SORTN) return;
  if (i < n) {
    float d0 = o1[2 * i + 0] - o2[2 * i + 0] + 1e-6f;
    float d1 = o1[2 * i + 1] - o2[2 * i + 1] + 1e-6f;
    sb[i] = sqrtf(d0 * d0 + d1 * d1);
  } else {
    sb[i] = -3.0e38f;
  }
}

// ---------- single-block bitonic sort, descending ----------
__global__ __launch_bounds__(1024)
void sort_kernel(float* __restrict__ sb) {
  for (int k = 2; k <= SORTN; k <<= 1) {
    for (int j = k >> 1; j > 0; j >>= 1) {
      for (int i = threadIdx.x; i < SORTN; i += 1024) {
        int ij = i ^ j;
        if (ij > i) {
          float a = sb[i], b = sb[ij];
          bool up = ((i & k) == 0);
          bool sw = up ? (a < b) : (a > b);
          if (sw) { sb[i] = b; sb[ij] = a; }
        }
      }
      __syncthreads();
    }
  }
}

// ---------- MLP head: fc1 -> LN -> relu -> fc2 -> LN -> relu -> fc3 -> sigmoid ----------
__global__ __launch_bounds__(512)
void head_kernel(const float* __restrict__ sb,
                 const float* __restrict__ f1w, const float* __restrict__ f1b,
                 const float* __restrict__ g1, const float* __restrict__ b1,
                 const float* __restrict__ f2w, const float* __restrict__ f2b,
                 const float* __restrict__ g2, const float* __restrict__ b2,
                 const float* __restrict__ f3w, const float* __restrict__ f3b,
                 float* __restrict__ out) {
  __shared__ float vals[TOPK];
  __shared__ float h1[128];
  __shared__ float h2[HDIM];
  __shared__ float red[512];
  __shared__ float stats[2];
  int t = threadIdx.x;
  for (int i = t; i < TOPK; i += 512) vals[i] = sb[i];
  __syncthreads();
  if (t < 128) {
    float a = f1b[t];
    for (int i = 0; i < TOPK; ++i) a += vals[i] * f1w[i * 128 + t];
    h1[t] = a;
  }
  __syncthreads();
  if (t == 0) {
    float s = 0.f, q = 0.f;
    for (int i = 0; i < 128; ++i) { s += h1[i]; q += h1[i] * h1[i]; }
    float m = s / 128.f;
    stats[0] = m;
    stats[1] = rsqrtf(q / 128.f - m * m + 1e-5f);
  }
  __syncthreads();
  if (t < 128) h1[t] = fmaxf((h1[t] - stats[0]) * stats[1] * g1[t] + b1[t], 0.0f);
  __syncthreads();
  {
    float a = f2b[t];
    for (int i = 0; i < 128; ++i) a += h1[i] * f2w[i * HDIM + t];
    h2[t] = a;
  }
  __syncthreads();
  if (t == 0) {
    float s = 0.f, q = 0.f;
    for (int i = 0; i < HDIM; ++i) { s += h2[i]; q += h2[i] * h2[i]; }
    float m = s / (float)HDIM;
    stats[0] = m;
    stats[1] = rsqrtf(q / (float)HDIM - m * m + 1e-5f);
  }
  __syncthreads();
  float hv = fmaxf((h2[t] - stats[0]) * stats[1] * g2[t] + b2[t], 0.0f);
  red[t] = hv * f3w[t];
  __syncthreads();
  for (int s = 256; s > 0; s >>= 1) {
    if (t < s) red[t] += red[t + s];
    __syncthreads();
  }
  if (t == 0) {
    float z = red[0] + f3b[0];
    out[0] = 1.0f / (1.0f + expf(-z));
  }
}

// ---------- host ----------
extern "C" void kernel_launch(void* const* d_in, const int* in_sizes, int n_in,
                              void* d_out, int out_size, void* d_ws, size_t ws_size,
                              hipStream_t stream) {
  const float* x1  = (const float*)d_in[0];
  const int*   ei1 = (const int*)  d_in[1];
  const float* x2  = (const float*)d_in[2];
  const int*   ei2 = (const int*)  d_in[3];
  const float* w11 = (const float*)d_in[4];  const float* b11 = (const float*)d_in[5];
  const float* w12 = (const float*)d_in[6];  const float* b12 = (const float*)d_in[7];
  const float* w21 = (const float*)d_in[8];  const float* b21 = (const float*)d_in[9];
  const float* w22 = (const float*)d_in[10]; const float* b22 = (const float*)d_in[11];
  const float* w31 = (const float*)d_in[12]; const float* b31 = (const float*)d_in[13];
  const float* w32 = (const float*)d_in[14]; const float* b32 = (const float*)d_in[15];
  const float* lw  = (const float*)d_in[16]; const float* lb  = (const float*)d_in[17];
  const float* f1w = (const float*)d_in[18]; const float* f1b = (const float*)d_in[19];
  const float* l1g = (const float*)d_in[20]; const float* l1b = (const float*)d_in[21];
  const float* f2w = (const float*)d_in[22]; const float* f2b = (const float*)d_in[23];
  const float* l2g = (const float*)d_in[24]; const float* l2b = (const float*)d_in[25];
  const float* f3w = (const float*)d_in[26]; const float* f3b = (const float*)d_in[27];

  char* ws = (char*)d_ws;
  size_t off = 0;
  auto alloc = [&](size_t bytes) -> void* {
    void* p = ws + off;
    off = (off + bytes + 255) & ~(size_t)255;
    return p;
  };
  unsigned short* wt11 = (unsigned short*)alloc((size_t)HDIM * FINDIM * 2);
  unsigned short* wt12 = (unsigned short*)alloc((size_t)HDIM * HDIM * 2);
  unsigned short* wt21 = (unsigned short*)alloc((size_t)HDIM * HDIM * 2);
  unsigned short* wt22 = (unsigned short*)alloc((size_t)HDIM * HDIM * 2);
  unsigned short* wt31 = (unsigned short*)alloc((size_t)HDIM * HDIM * 2);
  unsigned short* wt32 = (unsigned short*)alloc((size_t)HDIM * HDIM * 2);
  unsigned short* xb   = (unsigned short*)alloc((size_t)NNODES * FINDIM * 2);
  unsigned short* bufA = (unsigned short*)alloc((size_t)NNODES * HDIM * 2);
  unsigned short* bufB = (unsigned short*)alloc((size_t)NNODES * HDIM * 2);
  float* agg = (float*)alloc((size_t)NNODES * HDIM * 4);
  float* o1  = (float*)alloc((size_t)NNODES * 2 * 4);
  float* o2  = (float*)alloc((size_t)NNODES * 2 * 4);
  float* sb  = (float*)alloc((size_t)SORTN * 4);

  // weights -> transposed bf16 [Nout][K]
  wconv_kernel<<<(FINDIM * HDIM + 255) / 256, 256, 0, stream>>>(w11, wt11, FINDIM, HDIM);
  wconv_kernel<<<(HDIM * HDIM + 255) / 256, 256, 0, stream>>>(w12, wt12, HDIM, HDIM);
  wconv_kernel<<<(HDIM * HDIM + 255) / 256, 256, 0, stream>>>(w21, wt21, HDIM, HDIM);
  wconv_kernel<<<(HDIM * HDIM + 255) / 256, 256, 0, stream>>>(w22, wt22, HDIM, HDIM);
  wconv_kernel<<<(HDIM * HDIM + 255) / 256, 256, 0, stream>>>(w31, wt31, HDIM, HDIM);
  wconv_kernel<<<(HDIM * HDIM + 255) / 256, 256, 0, stream>>>(w32, wt32, HDIM, HDIM);

  dim3 gemm_grid((NNODES + 127) / 128, HDIM / 128);
  auto zero = [&](float* p, int n) {
    zero_kernel<<<(n + 255) / 256, 256, 0, stream>>>(p, n);
  };
  auto scatter = [&](const unsigned short* X, const int* s, const int* d, int F) {
    int total = NEDGES * (F >> 3);
    scatter_bf16_kernel<<<(total + 255) / 256, 256, 0, stream>>>(X, s, d, agg, NEDGES, F);
  };
  auto gemm = [&](const unsigned short* A, const float* A2, const unsigned short* WT,
                  const float* bias, unsigned short* C, int K) {
    gemm_bf16_kernel<<<gemm_grid, 256, 0, stream>>>(A, A2, WT, bias, C, NNODES, K, HDIM, 1);
  };

  auto run_graph = [&](const float* x, const int* ei, float* o) {
    const int* src = ei;
    const int* dst = ei + NEDGES;
    xconv_kernel<<<(NNODES * FINDIM + 255) / 256, 256, 0, stream>>>(x, xb, NNODES * FINDIM);
    // layer 1
    zero(agg, NNODES * FINDIM);
    scatter(xb, src, dst, FINDIM);
    gemm(xb, agg, wt11, b11, bufA, FINDIM);
    gemm(bufA, nullptr, wt12, b12, bufB, HDIM);
    // layer 2
    zero(agg, NNODES * HDIM);
    scatter(bufB, src, dst, HDIM);
    gemm(bufB, agg, wt21, b21, bufA, HDIM);
    gemm(bufA, nullptr, wt22, b22, bufB, HDIM);
    // layer 3
    zero(agg, NNODES * HDIM);
    scatter(bufB, src, dst, HDIM);
    gemm(bufB, agg, wt31, b31, bufA, HDIM);
    gemm(bufA, nullptr, wt32, b32, bufB, HDIM);
    // projection to 2 dims
    lin_kernel<<<(NNODES + 255) / 256, 256, 0, stream>>>(bufB, lw, lb, o, NNODES);
  };

  run_graph(x1, ei1, o1);
  run_graph(x2, ei2, o2);

  dist_kernel<<<(SORTN + 255) / 256, 256, 0, stream>>>(o1, o2, sb, NNODES);
  sort_kernel<<<1, 1024, 0, stream>>>(sb);
  head_kernel<<<1, 512, 0, stream>>>(sb, f1w, f1b, l1g, l1b, f2w, f2b, l2g, l2b,
                                     f3w, f3b, (float*)d_out);
}
